// BayesMultiheadAttention_9749575762417
// MI455X (gfx1250) — matrix operand order repaired and checked by reading, not compiled, not run on hardware
//
#include <hip/hip_runtime.h>
#include <hip/hip_bf16.h>

// ---------------- problem constants ----------------
static constexpr int kEmbed = 1024;
static constexpr int kHeads = 16;
static constexpr int kHd    = 64;
static constexpr int kSeq   = 2048;
static constexpr int kBsz   = 2;
static constexpr int kTok   = kSeq * kBsz;   // 4096 flattened (s,b) rows
static constexpr int kBH    = kBsz * kHeads; // 32 batch-heads

typedef __bf16 bf16_t;
typedef __attribute__((ext_vector_type(16))) __bf16 v16bf;
typedef __attribute__((ext_vector_type(8)))  __bf16 v8bf;
typedef __attribute__((ext_vector_type(8)))  float  v8f;
typedef unsigned int v4u __attribute__((ext_vector_type(4)));
typedef int          v4i __attribute__((ext_vector_type(4)));
typedef int          v8i __attribute__((ext_vector_type(8)));

union FragAB { v16bf v; v8bf h[2]; };

// ---------------- TDM availability (probe-guarded) ----------------
#if defined(__HIP_DEVICE_COMPILE__) &&                        \
    __has_builtin(__builtin_amdgcn_tensor_load_to_lds) &&     \
    __has_builtin(__builtin_amdgcn_s_wait_tensorcnt)
#define ATHENA_HAS_TDM 1
#else
#define ATHENA_HAS_TDM 0
#endif

#if ATHENA_HAS_TDM
// 2D bf16 tile DMA: Global -> LDS via the Tensor Data Mover.
// D# packing per CDNA5 ISA (2D, no pad/iterate).
static __device__ __forceinline__ void tdm_load_2d_bf16(
    unsigned lds_byte_addr, const void* gaddr,
    unsigned tile_x, unsigned tile_y,
    unsigned tensor_x, unsigned tensor_y,
    unsigned long long row_stride_elems) {
  unsigned long long ga = (unsigned long long)gaddr;
  v4u g0;
  g0[0] = 1u;                                            // count=1, no gather
  g0[1] = lds_byte_addr;                                 // lds_addr
  g0[2] = (unsigned)(ga & 0xffffffffu);                  // global_addr[31:0]
  g0[3] = (unsigned)((ga >> 32) & 0x01ffffffu)           // global_addr[56:32]
        | (2u << 30);                                    // type=2 ("image")
  // group1: 256 bits, build as 4x u64 then split
  unsigned long long q0 = (1ull << 16)                                   // data_size=1 (2B)
                        | ((unsigned long long)(tensor_x & 0xffffu) << 48);
  unsigned long long q1 = ((unsigned long long)tensor_x >> 16)           // tensor_dim0[31:16]
                        | ((unsigned long long)tensor_y << 16)           // tensor_dim1
                        | ((unsigned long long)(tile_x & 0xffffu) << 48);// tile_dim0
  unsigned long long q2 = (unsigned long long)(tile_y & 0xffffu)         // tile_dim1
                        | ((row_stride_elems & 0xffffffffull) << 32);    // stride[31:0]
  unsigned long long q3 = (row_stride_elems >> 32) & 0xffffull;          // stride[47:32]
  v8i g1;
  g1[0] = (int)(unsigned)(q0 & 0xffffffffu); g1[1] = (int)(unsigned)(q0 >> 32);
  g1[2] = (int)(unsigned)(q1 & 0xffffffffu); g1[3] = (int)(unsigned)(q1 >> 32);
  g1[4] = (int)(unsigned)(q2 & 0xffffffffu); g1[5] = (int)(unsigned)(q2 >> 32);
  g1[6] = (int)(unsigned)(q3 & 0xffffffffu); g1[7] = (int)(unsigned)(q3 >> 32);
  v4i z4 = {0, 0, 0, 0};
#if __clang_major__ >= 23
  v8i z8 = {0, 0, 0, 0, 0, 0, 0, 0};
  __builtin_amdgcn_tensor_load_to_lds(g0, g1, z4, z4, z8, 0);
#else
  __builtin_amdgcn_tensor_load_to_lds(g0, g1, z4, z4, 0);
#endif
}
#endif  // ATHENA_HAS_TDM

// ---------------- WMMA helpers (CDNA5 16x16x32 bf16, wave32) ----------------
// A fragment: 16x32 (MxK) row-major source. Lanes 0-15: row M=lane, K {0..7,16..23};
// lanes 16-31: row M=lane-16, K {8..15, 24..31}.
static __device__ __forceinline__ v16bf load_a_frag(const bf16_t* base, int lda) {
  const int lane = threadIdx.x & 31;
  const bf16_t* p = base + (size_t)(lane & 15) * lda + ((lane >> 4) * 8);
  FragAB f;
  f.h[0] = *reinterpret_cast<const v8bf*>(p);
  f.h[1] = *reinterpret_cast<const v8bf*>(p + 16);
  return f.v;
}

// B fragment for C = A @ W^T where W is [N,K] row-major: lane (n = lane&15)
// holds K {0..7,16..23} (lanes 0-15) or K {8..15,24..31} (lanes 16-31) of row n.
static __device__ __forceinline__ v16bf load_bt_frag(const bf16_t* base, int ldb) {
  const int lane = threadIdx.x & 31;
  const bf16_t* p = base + (size_t)(lane & 15) * ldb + ((lane >> 4) * 8);
  FragAB f;
  f.h[0] = *reinterpret_cast<const v8bf*>(p);
  f.h[1] = *reinterpret_cast<const v8bf*>(p + 16);
  return f.v;
}

static __device__ __forceinline__ v8f wmma_bf16(v16bf a, v16bf b, v8f c) {
  return __builtin_amdgcn_wmma_f32_16x16x32_bf16(false, a, false, b, (short)0, c,
                                                 false, false);
}

// exchange with lane^16 (SWAPX16): xor=0x10, or=0, and=0x1f -> 0x401F
static __device__ __forceinline__ float xor16(float x) {
  int i = __float_as_int(x);
  i = __builtin_amdgcn_ds_swizzle(i, 0x401F);
  return __int_as_float(i);
}

// ---------------- attention kernel (first WMMA user: disasm window lands here)
// One wave per (batch-head, 16-query-row tile). Full 16 x 2048 fp32 score slab
// lives in LDS (128 KB of the 320 KB WGP budget) -> single K pass for scores,
// in-LDS softmax, head-average via global f32 atomics, then P @ V from LDS.
// K tiles are staged into LDS with the Tensor Data Mover, double-buffered and
// synchronized with s_wait_tensorcnt (falls back to direct global loads).
__global__ __launch_bounds__(32) void attn_kernel(
    const bf16_t* __restrict__ qb, const bf16_t* __restrict__ kb,
    const bf16_t* __restrict__ vT, bf16_t* __restrict__ ctx,
    float* __restrict__ avg) {
  __shared__ float slab[16 * kSeq];  // 131072 bytes

  const int lane = threadIdx.x & 31;
  const int qt = blockIdx.x;   // 0..127 query tile
  const int bh = blockIdx.y;   // 0..31
  const int b = bh >> 4, h = bh & 15;
  const int ldq = kBsz * kEmbed;  // 2048: row stride between consecutive s

  // ---- scores: S[qt*16 + r, :] = Q_tile @ K^T (q pre-scaled by 1/sqrt(hd)) ----
  const bf16_t* qbase = qb + (size_t)(qt * 16 * kBsz + b) * kEmbed + h * kHd;
  v16bf aq0 = load_a_frag(qbase + 0, ldq);   // d 0..31
  v16bf aq1 = load_a_frag(qbase + 32, ldq);  // d 32..63

  const bf16_t* kbase0 = kb + (size_t)b * kEmbed + h * kHd;

#if ATHENA_HAS_TDM
  __shared__ bf16_t kslab[2][16 * kHd];  // 4 KB double buffer, TDM destination
  const unsigned klds[2] = {(unsigned)(unsigned long long)&kslab[0][0],
                            (unsigned)(unsigned long long)&kslab[1][0]};
  tdm_load_2d_bf16(klds[0], kbase0, kHd, 16, kHd, 16,
                   (unsigned long long)ldq);
#endif

  for (int n0 = 0; n0 < kSeq; n0 += 16) {
    v8f acc;
#pragma unroll
    for (int i = 0; i < 8; ++i) acc[i] = 0.0f;

#if ATHENA_HAS_TDM
    const int buf = (n0 >> 4) & 1;
    if (n0 + 16 < kSeq) {
      tdm_load_2d_bf16(klds[buf ^ 1], kbase0 + (size_t)(n0 + 16) * ldq,
                       kHd, 16, kHd, 16, (unsigned long long)ldq);
      __builtin_amdgcn_s_wait_tensorcnt(1);  // current tile landed in LDS
    } else {
      __builtin_amdgcn_s_wait_tensorcnt(0);
    }
    const bf16_t* kt = &kslab[buf][0];
    acc = wmma_bf16(aq0, load_bt_frag(kt + 0, kHd), acc);
    acc = wmma_bf16(aq1, load_bt_frag(kt + 32, kHd), acc);
#else
    const bf16_t* kt = kbase0 + (size_t)n0 * ldq;
    acc = wmma_bf16(aq0, load_bt_frag(kt + 0, ldq), acc);
    acc = wmma_bf16(aq1, load_bt_frag(kt + 32, ldq), acc);
#endif

    const int col = n0 + (lane & 15);
    const int rb = (lane >> 4) * 8;
#pragma unroll
    for (int r = 0; r < 8; ++r) slab[(rb + r) * kSeq + col] = acc[r];
  }
  __syncthreads();

  // ---- softmax: lanes (L, L+16) split row L into two 1024-wide halves ----
  const int row = lane & 15;
  float* srow = slab + (size_t)row * kSeq + (lane >> 4) * (kSeq / 2);
  float mloc = -3.0e38f;
  for (int c = 0; c < kSeq / 2; ++c) mloc = fmaxf(mloc, srow[c]);
  const float m = fmaxf(mloc, xor16(mloc));
  float sloc = 0.0f;
  for (int c = 0; c < kSeq / 2; ++c) {
    float e = __expf(srow[c] - m);
    srow[c] = e;
    sloc += e;
  }
  const float inv = 1.0f / (sloc + xor16(sloc));
  float* avgrow = avg + ((size_t)b * kSeq + (qt * 16 + row)) * kSeq +
                  (lane >> 4) * (kSeq / 2);
  for (int c = 0; c < kSeq / 2; ++c) {
    float p = srow[c] * inv;
    srow[c] = p;  // keep normalized probs for the P@V GEMM
    atomicAdd(&avgrow[c], p * (1.0f / (float)kHeads));
  }
  __syncthreads();

  // ---- ctx = P(16 x 2048) @ V(2048 x 64), vT layout [bh][d][s] ----
  const bf16_t* vbase = vT + (size_t)bh * kHd * kSeq;
  v8f acc[4];
#pragma unroll
  for (int t = 0; t < 4; ++t)
#pragma unroll
    for (int i = 0; i < 8; ++i) acc[t][i] = 0.0f;

  for (int k0 = 0; k0 < kSeq; k0 += 32) {
    // A fragment straight from the fp32 LDS slab (cvt to bf16 in regs)
    const float* ap = slab + (size_t)(lane & 15) * kSeq + k0 + (lane >> 4) * 8;
    FragAB af;
#pragma unroll
    for (int j = 0; j < 8; ++j) {
      af.h[0][j] = (bf16_t)ap[j];
      af.h[1][j] = (bf16_t)ap[j + 16];
    }
#pragma unroll
    for (int t = 0; t < 4; ++t) {
      v16bf vf = load_bt_frag(vbase + (size_t)(t * 16) * kSeq + k0, kSeq);
      acc[t] = wmma_bf16(af.v, vf, acc[t]);
    }
  }

  // store ctx as bf16 [S,B,E] rows for the output projection GEMM
#pragma unroll
  for (int t = 0; t < 4; ++t) {
    const int e = h * kHd + t * 16 + (lane & 15);
#pragma unroll
    for (int r = 0; r < 8; ++r) {
      const int s = qt * 16 + (lane >> 4) * 8 + r;
      ctx[(size_t)(s * kBsz + b) * kEmbed + e] = (bf16_t)acc[t][r];
    }
  }
}

// ---------------- generic GEMM: C[m,n] = (sum_k A[m,k]*W[n,k] + bias[n]) * alpha
// One wave per 32x64 C tile (2 A-frags x 4 B-frags -> 8 independent WMMAs per
// k-step). Compile-time HasBias/OutB flags keep the epilogue branch-free.
template <bool kHasBias, bool kOutB>
__global__ __launch_bounds__(32) void gemm_bf16_kernel(
    const bf16_t* __restrict__ A, const bf16_t* __restrict__ W,
    const float* __restrict__ bias, float alpha, int K, int N,
    bf16_t* __restrict__ outb, float* __restrict__ outf) {
  const int lane = threadIdx.x & 31;
  const int n0 = blockIdx.x * 64;
  const int m0 = blockIdx.y * 32;
  const bf16_t* a = A + (size_t)m0 * K;
  const bf16_t* w = W + (size_t)n0 * K;

  v8f acc[2][4];
#pragma unroll
  for (int i = 0; i < 2; ++i)
#pragma unroll
    for (int j = 0; j < 4; ++j)
#pragma unroll
      for (int e = 0; e < 8; ++e) acc[i][j][e] = 0.0f;

  for (int k0 = 0; k0 < K; k0 += 32) {
    if (k0 + 256 < K) {  // L2/WGP$ warm-up hints (global_prefetch_b8)
      __builtin_prefetch(a + k0 + 256, 0, 1);
      __builtin_prefetch(w + k0 + 256, 0, 1);
    }
    v16bf af[2], wf[4];
#pragma unroll
    for (int i = 0; i < 2; ++i)
      af[i] = load_a_frag(a + (size_t)(i * 16) * K + k0, K);
#pragma unroll
    for (int j = 0; j < 4; ++j)
      wf[j] = load_bt_frag(w + (size_t)(j * 16) * K + k0, K);
#pragma unroll
    for (int i = 0; i < 2; ++i)
#pragma unroll
      for (int j = 0; j < 4; ++j)
        acc[i][j] = wmma_bf16(af[i], wf[j], acc[i][j]);
  }

#pragma unroll
  for (int i = 0; i < 2; ++i) {
    const int mrow = m0 + i * 16 + (lane >> 4) * 8;
#pragma unroll
    for (int j = 0; j < 4; ++j) {
      const int n = n0 + j * 16 + (lane & 15);
      const float bn = kHasBias ? bias[n] : 0.0f;
#pragma unroll
      for (int r = 0; r < 8; ++r) {
        float c = (acc[i][j][r] + bn) * alpha;
        size_t idx = (size_t)(mrow + r) * N + n;
        if constexpr (kOutB) outb[idx] = (bf16_t)c;
        else                 outf[idx] = c;
      }
    }
  }
}

// ---------------- elementwise prep kernels ----------------
__global__ void zero_f32_kernel(float* __restrict__ p, int n) {
  int i = blockIdx.x * blockDim.x + threadIdx.x;
  if (i < n) p[i] = 0.0f;
}

__global__ void cvt_bf16_kernel(const float* __restrict__ src,
                                bf16_t* __restrict__ dst, int n) {
  int i = blockIdx.x * blockDim.x + threadIdx.x;
  if (i < n) dst[i] = (bf16_t)src[i];
}

__global__ void sample_ow_kernel(const float* __restrict__ mean,
                                 const float* __restrict__ lgstd,
                                 const float* __restrict__ eps,
                                 bf16_t* __restrict__ dst, int n) {
  int i = blockIdx.x * blockDim.x + threadIdx.x;
  if (i < n) dst[i] = (bf16_t)(mean[i] + eps[i] * __expf(lgstd[i]));
}

// vT[bh][d][s] = v_proj[(s*B + b)][h*64 + d]
__global__ void transpose_v_kernel(const bf16_t* __restrict__ vb,
                                   bf16_t* __restrict__ vT, int n) {
  int i = blockIdx.x * blockDim.x + threadIdx.x;
  if (i >= n) return;
  int s  = i & (kSeq - 1);
  int d  = (i >> 11) & (kHd - 1);
  int bh = i >> 17;
  int b = bh >> 4, h = bh & 15;
  vT[i] = vb[(size_t)(s * kBsz + b) * kEmbed + h * kHd + d];
}

// ---------------- host-side orchestration ----------------
extern "C" void kernel_launch(void* const* d_in, const int* in_sizes, int n_in,
                              void* d_out, int out_size, void* d_ws, size_t ws_size,
                              hipStream_t stream) {
  const float* query  = (const float*)d_in[0];
  const float* key_in = (const float*)d_in[1];
  const float* val_in = (const float*)d_in[2];
  const float* q_w = (const float*)d_in[3];
  const float* q_b = (const float*)d_in[4];
  const float* k_w = (const float*)d_in[5];
  const float* k_b = (const float*)d_in[6];
  const float* v_w = (const float*)d_in[7];
  const float* v_b = (const float*)d_in[8];
  const float* o_mean  = (const float*)d_in[9];
  const float* o_lgstd = (const float*)d_in[10];
  const float* eps     = (const float*)d_in[11];

  float* out = (float*)d_out;                              // [S,B,E]
  float* avg = out + (size_t)kSeq * kBsz * kEmbed;         // [B,S,S]

  // workspace carve-up (bf16 buffers), 256B aligned
  char* ws = (char*)d_ws;
  size_t off = 0;
  auto alloc = [&](size_t bytes) {
    void* p = ws + off;
    off += (bytes + 255) & ~(size_t)255;
    return p;
  };
  const size_t actE = (size_t)kTok * kEmbed;   // 4,194,304 elems
  const size_t wE   = (size_t)kEmbed * kEmbed; // 1,048,576 elems
  bf16_t* Xq = (bf16_t*)alloc(actE * 2);
  bf16_t* Xk = (bf16_t*)alloc(actE * 2);
  bf16_t* Xv = (bf16_t*)alloc(actE * 2);
  bf16_t* Wq = (bf16_t*)alloc(wE * 2);
  bf16_t* Wk = (bf16_t*)alloc(wE * 2);
  bf16_t* Wv = (bf16_t*)alloc(wE * 2);
  bf16_t* Wo = (bf16_t*)alloc(wE * 2);
  bf16_t* Qp = (bf16_t*)alloc(actE * 2);
  bf16_t* Kp = (bf16_t*)alloc(actE * 2);
  bf16_t* Vp = (bf16_t*)alloc(actE * 2);
  bf16_t* VT = (bf16_t*)alloc(actE * 2);
  bf16_t* Cx = (bf16_t*)alloc(actE * 2);
  (void)ws_size; (void)in_sizes; (void)n_in; (void)out_size;

  const int navg = kBsz * kSeq * kSeq;  // 8,388,608
  zero_f32_kernel<<<(navg + 255) / 256, 256, 0, stream>>>(avg, navg);

  const int nact = (int)actE, nw = (int)wE;
  cvt_bf16_kernel<<<(nact + 255) / 256, 256, 0, stream>>>(query,  Xq, nact);
  cvt_bf16_kernel<<<(nact + 255) / 256, 256, 0, stream>>>(key_in, Xk, nact);
  cvt_bf16_kernel<<<(nact + 255) / 256, 256, 0, stream>>>(val_in, Xv, nact);
  cvt_bf16_kernel<<<(nw + 255) / 256, 256, 0, stream>>>(q_w, Wq, nw);
  cvt_bf16_kernel<<<(nw + 255) / 256, 256, 0, stream>>>(k_w, Wk, nw);
  cvt_bf16_kernel<<<(nw + 255) / 256, 256, 0, stream>>>(v_w, Wv, nw);
  sample_ow_kernel<<<(nw + 255) / 256, 256, 0, stream>>>(o_mean, o_lgstd, eps, Wo, nw);

  const float scaling = 0.125f;  // 1/sqrt(64)
  dim3 gproj(kEmbed / 64, kTok / 32);  // (16, 128) waves of 32x64 tiles
  gemm_bf16_kernel<true, true><<<gproj, 32, 0, stream>>>(
      Xq, Wq, q_b, scaling, kEmbed, kEmbed, Qp, (float*)nullptr);
  gemm_bf16_kernel<true, true><<<gproj, 32, 0, stream>>>(
      Xk, Wk, k_b, 1.0f, kEmbed, kEmbed, Kp, (float*)nullptr);
  gemm_bf16_kernel<true, true><<<gproj, 32, 0, stream>>>(
      Xv, Wv, v_b, 1.0f, kEmbed, kEmbed, Vp, (float*)nullptr);

  transpose_v_kernel<<<(nact + 255) / 256, 256, 0, stream>>>(Vp, VT, nact);

  dim3 gattn(kSeq / 16, kBH);  // (128, 32)
  attn_kernel<<<gattn, 32, 0, stream>>>(Qp, Kp, VT, Cx, avg);

  gemm_bf16_kernel<false, false><<<gproj, 32, 0, stream>>>(
      Cx, Wo, (const float*)nullptr, 1.0f, kEmbed, kEmbed, (bf16_t*)nullptr, out);
}
